// ScaledDotProductAttention_26036091748419
// MI455X (gfx1250) — matrix operand order repaired and checked by reading, not compiled, not run on hardware
//
#include <hip/hip_runtime.h>
#include <math.h>

// ---------------- problem constants ----------------
#define B_   32
#define D_   8192
#define DK_  128
#define DV_  128
#define NSG  8                         // workgroups per batch
#define SPLITS 16                      // key splits per (batch, m-half) = NSG * 2 pairs
#define KEYS_PER_SPLIT (D_ / SPLITS)   // 512
#define KB_  32                        // keys per inner iteration
#define ITERS (KEYS_PER_SPLIT / KB_)   // 16
#define KPAD 132                       // LDS row stride (floats) for 32 x 128 tiles
#define PPAD 36                        // LDS row stride (bf16) for P scratch
#define PART (16*DV_ + 32)             // floats per wave partial: O(16x128), m(16), l(16)
#define SCL  (0.08838834764831845f * 1.44269504088896340f) // (1/sqrt(128)) * log2(e)

typedef __attribute__((ext_vector_type(16))) __bf16 v16bf;
typedef __attribute__((ext_vector_type(8)))  float  v8f;
typedef __attribute__((ext_vector_type(4)))  unsigned int v4u;
typedef __attribute__((ext_vector_type(8)))  int v8i;
typedef __attribute__((ext_vector_type(4)))  int v4i;

#if __has_builtin(__builtin_amdgcn_tensor_load_to_lds)
#define USE_TDM 1
#else
#define USE_TDM 0
#warning "tensor_load_to_lds builtin NOT available - using sync float4 staging fallback"
#endif

#if !__has_builtin(__builtin_amdgcn_s_wait_tensorcnt)
#warning "s_wait_tensorcnt builtin NOT available"
#endif

#if USE_TDM
// DMA one 32-row x 128-float f32 tile (row-major, row stride 128 floats in memory)
// into LDS with a 132-float row stride (pad_interval=128 DW, pad_amount=4 DW).
__device__ __forceinline__ void tdm_load_tile32(const float* g, unsigned lds_byte)
{
  unsigned long long ga = (unsigned long long)g;
  unsigned galo = (unsigned)__builtin_amdgcn_readfirstlane((int)(ga & 0xffffffffull));
  unsigned gahi = (unsigned)__builtin_amdgcn_readfirstlane((int)(ga >> 32));
  unsigned la   = (unsigned)__builtin_amdgcn_readfirstlane((int)lds_byte);
  v4u g0;
  g0[0] = 1u;                                   // count = 1 (valid descriptor)
  g0[1] = la;                                   // lds_addr (bytes)
  g0[2] = galo;                                 // global_addr[31:0]
  g0[3] = (gahi & 0x01ffffffu) | (2u << 30);    // global_addr[56:32] | type=2 (image)
  v8i g1;
  g1[0] = (int)((2u<<16) | (1u<<20) | (6u<<22) | (3u<<25)); // data_size=4B, pad_en, interval=128DW, amount=4DW
  g1[1] = (int)(((unsigned)DK_) << 16);         // tensor_dim0 = 128   (bits 79:48)
  g1[2] = (int)(((unsigned)KB_) << 16);         // tensor_dim1 = 32    (bits 111:80)
  g1[3] = (int)(((unsigned)DK_) << 16);         // tile_dim0   = 128   (bits 127:112)
  g1[4] = (int)KB_;                              // tile_dim1 = 32, tile_dim2 = 0
  g1[5] = (int)DK_;                              // tensor_dim0_stride = 128
  g1[6] = 0;
  g1[7] = 0;
  v4i z4 = {0, 0, 0, 0};
#if __clang_major__ >= 23
  v8i z8 = {0, 0, 0, 0, 0, 0, 0, 0};
  __builtin_amdgcn_tensor_load_to_lds(g0, g1, z4, z4, z8, 0);
#else
  __builtin_amdgcn_tensor_load_to_lds(g0, g1, z4, z4, 0);
#endif
}
#endif

// ---------------------------------------------------------------------------
// Kernel 1: flash-attention partials.
// grid = B*NSG blocks, 128 threads (4 waves = 2 pairs).
// Pair p handles key range [split*512, split*512+512); wave mh in {0,1} of the
// pair handles query rows mh*16 .. mh*16+15 and shares the pair's LDS K/V tiles.
// ---------------------------------------------------------------------------
__global__ __launch_bounds__(128) void fa_partial_kernel(
    const float* __restrict__ query, const float* __restrict__ keys,
    const float* __restrict__ values, float* __restrict__ ws)
{
  __shared__ __align__(16) float  sK[2][2][32 * KPAD];
  __shared__ __align__(16) float  sV[2][2][32 * KPAD];
  __shared__ __align__(16) __bf16 sP[4][16 * PPAD];

  const int tid   = threadIdx.x;
  const int lane  = tid & 31;
  const int wave  = tid >> 5;
  const int pair  = wave >> 1;
  const int mh    = wave & 1;
  const int b     = blockIdx.x / NSG;
  const int sg    = blockIdx.x % NSG;
  const int split = sg * 2 + pair;
  const int key0  = split * KEYS_PER_SPLIT;
  const int c     = lane & 15;   // A-row / B-col / C-col index
  const int h     = lane >> 4;   // lane half

  // ---- Q fragments (bf16 A-layout, 16x32 per K-tile), rows mh*16 + (0..15)
  const float* qrow = query + (size_t)(mh * 16 + c) * DK_;
  v16bf aq[4];
#pragma unroll
  for (int kt = 0; kt < 4; ++kt) {
#pragma unroll
    for (int v = 0; v < 8; ++v) {
      int K = kt * 32 + h * 16 + 2 * v;
      aq[kt][2 * v]     = (__bf16)qrow[K];
      aq[kt][2 * v + 1] = (__bf16)qrow[K + 1];
    }
  }

  // wave mh==0 stages K tiles, wave mh==1 stages V tiles (shared by the pair)
  const float* gsrc_base = (mh == 0) ? (keys   + ((size_t)b * D_ + key0) * DK_)
                                     : (values + ((size_t)b * D_ + key0) * DK_);
  float* ldst0 = (mh == 0) ? &sK[pair][0][0] : &sV[pair][0][0];
  float* ldst1 = (mh == 0) ? &sK[pair][1][0] : &sV[pair][1][0];

  auto stage = [&](int it, int db) {
    const float* g = gsrc_base + (size_t)it * KB_ * DK_;
    float* dst = db ? ldst1 : ldst0;
#if USE_TDM
    (void)lane;
    tdm_load_tile32(g, (unsigned)(unsigned long long)dst);
#else
#pragma unroll 4
    for (int i = 0; i < 32; ++i) {
      float4 x = *(const float4*)(g + i * DK_ + 4 * lane);
      *(float4*)(dst + i * KPAD + 4 * lane) = x;
    }
#endif
  };

  // ---- accumulator state
  float m2[8], lsum[8];
  v8f Oa[8];
  const v8f vz = {0.f, 0.f, 0.f, 0.f, 0.f, 0.f, 0.f, 0.f};
#pragma unroll
  for (int r = 0; r < 8; ++r) { m2[r] = -__builtin_inff(); lsum[r] = 0.f; }
#pragma unroll
  for (int t = 0; t < 8; ++t) Oa[t] = vz;

  stage(0, 0);
  __bf16* Pw = &sP[wave][0];

  for (int it = 0; it < ITERS; ++it) {
    const int db = it & 1;
#if USE_TDM
    __builtin_amdgcn_s_wait_tensorcnt(0);
#endif
    __syncthreads();
    if (it + 1 < ITERS) stage(it + 1, db ^ 1);

    const float* K0 = &sK[pair][db][0];
    const float* V0 = &sV[pair][db][0];

    // ---- S = Q * K^T for 32 keys -> two 16x16 f32 tiles (8 WMMAs)
    v8f s0 = vz, s1 = vz;
#pragma unroll
    for (int kt = 0; kt < 4; ++kt) {
      v16bf bk0, bk1;
      const float* p0 = K0 + c * KPAD + kt * 32 + h * 16;          // keys 0..15
      const float* p1 = K0 + (c + 16) * KPAD + kt * 32 + h * 16;   // keys 16..31
#pragma unroll
      for (int j = 0; j < 8; ++j) {
        bk0[2 * j]     = (__bf16)p0[2 * j];
        bk0[2 * j + 1] = (__bf16)p0[2 * j + 1];
        bk1[2 * j]     = (__bf16)p1[2 * j];
        bk1[2 * j + 1] = (__bf16)p1[2 * j + 1];
      }
      s0 = __builtin_amdgcn_wmma_f32_16x16x32_bf16(false, aq[kt], false, bk0, (short)0, s0, false, false);
      s1 = __builtin_amdgcn_wmma_f32_16x16x32_bf16(false, aq[kt], false, bk1, (short)0, s1, false, false);
    }

    // ---- online softmax (rows r+8h, cols striped over 16 lanes per half)
    float rm[8], al[8], rs[8];
#pragma unroll
    for (int r = 0; r < 8; ++r) rm[r] = fmaxf(s0[r], s1[r]);
#pragma unroll
    for (int msk = 1; msk < 16; msk <<= 1) {
#pragma unroll
      for (int r = 0; r < 8; ++r) rm[r] = fmaxf(rm[r], __shfl_xor(rm[r], msk, 32));
    }
#pragma unroll
    for (int r = 0; r < 8; ++r) {
      float mn = fmaxf(m2[r], rm[r]);
      al[r] = exp2f((m2[r] - mn) * SCL);
      float p0v = exp2f((s0[r] - mn) * SCL);
      float p1v = exp2f((s1[r] - mn) * SCL);
      s0[r] = p0v; s1[r] = p1v;
      rs[r] = p0v + p1v;
      m2[r] = mn;
    }
#pragma unroll
    for (int msk = 1; msk < 16; msk <<= 1) {
#pragma unroll
      for (int r = 0; r < 8; ++r) rs[r] += __shfl_xor(rs[r], msk, 32);
    }
#pragma unroll
    for (int r = 0; r < 8; ++r) lsum[r] = lsum[r] * al[r] + rs[r];
#pragma unroll
    for (int t = 0; t < 8; ++t) {
#pragma unroll
      for (int r = 0; r < 8; ++r) Oa[t][r] *= al[r];
    }

    // ---- re-layout P: C-layout (16x32 f32) -> A-layout (bf16) via LDS bounce
#pragma unroll
    for (int r = 0; r < 8; ++r) {
      Pw[(8 * h + r) * PPAD + c]      = (__bf16)s0[r];
      Pw[(8 * h + r) * PPAD + c + 16] = (__bf16)s1[r];
    }
    asm volatile("" ::: "memory");   // same-wave DS ops are in-order; stop compiler reordering
    v16bf pa;
#pragma unroll
    for (int v = 0; v < 8; ++v) {
      int C = h * 16 + 2 * v;
      pa[2 * v]     = Pw[c * PPAD + C];
      pa[2 * v + 1] = Pw[c * PPAD + C + 1];
    }

    // ---- O += P(16x32) * V(32x128): 8 WMMAs
#pragma unroll
    for (int t = 0; t < 8; ++t) {
      v16bf bv;
#pragma unroll
      for (int j = 0; j < 8; ++j) {
#pragma unroll
        for (int e = 0; e < 2; ++e) {
          int key = h * 16 + 2 * j + e;
          bv[2 * j + e] = (__bf16)V0[key * KPAD + t * 16 + c];
        }
      }
      Oa[t] = __builtin_amdgcn_wmma_f32_16x16x32_bf16(false, pa, false, bv, (short)0, Oa[t], false, false);
    }
  }

  // ---- write per-wave partial: O(16x128), m(16), l(16)
  float* pb = ws + (size_t)((b * 2 + mh) * SPLITS + split) * PART;
#pragma unroll
  for (int t = 0; t < 8; ++t) {
#pragma unroll
    for (int r = 0; r < 8; ++r)
      pb[(8 * h + r) * DV_ + t * 16 + c] = Oa[t][r];
  }
  if (c == 0) {
#pragma unroll
    for (int r = 0; r < 8; ++r) {
      pb[16 * DV_ + 8 * h + r]      = m2[r];
      pb[16 * DV_ + 16 + 8 * h + r] = lsum[r];
    }
  }
}

// ---------------------------------------------------------------------------
// Kernel 2: merge the 16 split-softmax partials per (batch, m-half).
// grid = B*2 blocks, 128 threads (thread = dv index).
// ---------------------------------------------------------------------------
__global__ __launch_bounds__(128) void fa_combine_kernel(
    const float* __restrict__ ws, float* __restrict__ out)
{
  const int b  = blockIdx.x >> 1;
  const int mh = blockIdx.x & 1;
  const int v  = threadIdx.x;  // 0..127
  const float* base = ws + (size_t)((b * 2 + mh) * SPLITS) * PART;

  for (int row = 0; row < 16; ++row) {
    float M = -__builtin_inff();
    float ms[SPLITS], ls[SPLITS];
#pragma unroll
    for (int s = 0; s < SPLITS; ++s) {
      ms[s] = base[(size_t)s * PART + 16 * DV_ + row];
      ls[s] = base[(size_t)s * PART + 16 * DV_ + 16 + row];
      M = fmaxf(M, ms[s]);
    }
    float L = 0.f, acc = 0.f;
#pragma unroll
    for (int s = 0; s < SPLITS; ++s) {
      float w = exp2f((ms[s] - M) * SCL);
      L   += w * ls[s];
      acc += w * base[(size_t)s * PART + row * DV_ + v];
    }
    out[((size_t)b * 32 + mh * 16 + row) * DV_ + v] = acc / L;
  }
}

// ---------------------------------------------------------------------------
extern "C" void kernel_launch(void* const* d_in, const int* in_sizes, int n_in,
                              void* d_out, int out_size, void* d_ws, size_t ws_size,
                              hipStream_t stream)
{
  (void)in_sizes; (void)n_in; (void)out_size; (void)ws_size;
  const float* query  = (const float*)d_in[0];
  const float* keys   = (const float*)d_in[1];
  const float* values = (const float*)d_in[2];
  float* out = (float*)d_out;
  float* ws  = (float*)d_ws;   // needs 32*2*16*PART*4 = ~8.5 MB

  fa_partial_kernel<<<dim3(B_ * NSG), dim3(128), 0, stream>>>(query, keys, values, ws);
  fa_combine_kernel<<<dim3(B_ * 2), dim3(DV_), 0, stream>>>(ws, out);
}
